// Decoder_44444321579015
// MI455X (gfx1250) — matrix operand order repaired and checked by reading, not compiled, not run on hardware
//
#include <hip/hip_runtime.h>
#include <stdint.h>

typedef __attribute__((ext_vector_type(16))) _Float16 v16h;
typedef __attribute__((ext_vector_type(8)))  _Float16 v8h;
typedef __attribute__((ext_vector_type(8)))  float    v8f;

#define B_     16
#define NMELS  80
#define TDEC   200
#define TENC   256
#define ENC_   512
#define PRE_   256
#define ARNN_  1024
#define DEC_   1024
#define ADIM_  128
#define NFILT_ 32
#define KCONV  31

// ---------------------------------------------------------------- utilities
__device__ __forceinline__ float sigmf(float x) { return 1.0f / (1.0f + __expf(-x)); }

__device__ __forceinline__ unsigned hash_u32(unsigned x) {
  x ^= x >> 16; x *= 0x7feb352du;
  x ^= x >> 15; x *= 0x846ca68bu;
  x ^= x >> 16; return x;
}

// Load one A fragment (16x32 f16 tile slice for this lane), CDNA5 ISA 7.12.2:
// lanes 0-15 row M=lane, K {0..7,16..23}; lanes 16-31 K {8..15,24..31}
__device__ __forceinline__ v16h load_afrag(const _Float16* __restrict__ Ap, int k0) {
  v8h alo = *(const v8h*)(Ap + k0);
  v8h ahi = *(const v8h*)(Ap + k0 + 16);
  return __builtin_shufflevector(alo, ahi, 0,1,2,3,4,5,6,7,8,9,10,11,12,13,14,15);
}
// Load one B fragment: lane%16 = N col, K {0..7,16..23} (lanes 0-15) / {8..15,24..31} (lanes 16-31)
__device__ __forceinline__ v16h load_bfrag(const _Float16* __restrict__ Wp, int k0) {
  v8h blo = *(const v8h*)(Wp + k0);
  v8h bhi = *(const v8h*)(Wp + k0 + 16);
  return __builtin_shufflevector(blo, bhi, 0,1,2,3,4,5,6,7,8,9,10,11,12,13,14,15);
}

// One K-segment of a single 16x16 WMMA tile (generic GEMM path).
__device__ __forceinline__ void wmma_seg(v8f& acc,
                                         const _Float16* __restrict__ A, int lda, int K,
                                         const _Float16* __restrict__ Wrow, int koff,
                                         int arow, int hi)
{
  if (K == 0) return;
  const _Float16* Ap = A + (long long)arow * lda + hi * 8;
  const _Float16* Wp = Wrow + koff + hi * 8;
  for (int k0 = 0; k0 < K; k0 += 32) {
    v16h a = load_afrag(Ap, k0);
    v16h b = load_bfrag(Wp, k0);
    acc = __builtin_amdgcn_wmma_f32_16x16x32_f16(false, a, false, b, (short)0, acc,
                                                 false, false);
  }
}

// One K-segment feeding FOUR gate tiles from a single A fragment load.
__device__ __forceinline__ void wmma_seg4(v8f& g0, v8f& g1, v8f& g2, v8f& g3,
                                          const _Float16* __restrict__ A, int lda, int K,
                                          const _Float16* __restrict__ w0,
                                          const _Float16* __restrict__ w1,
                                          const _Float16* __restrict__ w2,
                                          const _Float16* __restrict__ w3,
                                          int koff, int arow, int hi)
{
  if (K == 0) return;
  const _Float16* Ap = A + (long long)arow * lda + hi * 8;
  const int bo = koff + hi * 8;
  for (int k0 = 0; k0 < K; k0 += 32) {
    v16h a = load_afrag(Ap, k0);
    v16h b0 = load_bfrag(w0 + bo, k0);
    v16h b1 = load_bfrag(w1 + bo, k0);
    v16h b2 = load_bfrag(w2 + bo, k0);
    v16h b3 = load_bfrag(w3 + bo, k0);
    g0 = __builtin_amdgcn_wmma_f32_16x16x32_f16(false, a, false, b0, (short)0, g0, false, false);
    g1 = __builtin_amdgcn_wmma_f32_16x16x32_f16(false, a, false, b1, (short)0, g1, false, false);
    g2 = __builtin_amdgcn_wmma_f32_16x16x32_f16(false, a, false, b2, (short)0, g2, false, false);
    g3 = __builtin_amdgcn_wmma_f32_16x16x32_f16(false, a, false, b3, (short)0, g3, false, false);
  }
}

// ---------------------------------------------------------------- generic GEMM
// C[M,N] = concat_K(A0|A1|A2) * W[N,Ktot]^T + bias, strided C output.
__global__ __launch_bounds__(256) void gemm16_multi(
    const _Float16* __restrict__ A0, int lda0, int K0,
    const _Float16* __restrict__ A1, int lda1, int K1,
    const _Float16* __restrict__ A2, int lda2, int K2,
    const _Float16* __restrict__ W, int ldw,
    float* __restrict__ C, long long ldc, long long cstride,
    int M, int N, const float* __restrict__ bias)
{
  const int wave = (int)((blockIdx.x * blockDim.x + threadIdx.x) >> 5);
  const int lane = threadIdx.x & 31;
  const int ntiles = N >> 4;
  const int mtiles = (M + 15) >> 4;
  if (wave >= mtiles * ntiles) return;          // wave-uniform: EXEC all-1 for WMMA
  const int tm = wave / ntiles;
  const int tn = wave - tm * ntiles;
  const int hi  = lane >> 4;
  const int l15 = lane & 15;
  const int arow = tm * 16 + l15;
  const _Float16* Wrow = W + (long long)(tn * 16 + l15) * ldw;

  v8f acc = {0.f, 0.f, 0.f, 0.f, 0.f, 0.f, 0.f, 0.f};
  wmma_seg(acc, A0, lda0, K0, Wrow, 0,        arow, hi);
  wmma_seg(acc, A1, lda1, K1, Wrow, K0,       arow, hi);
  wmma_seg(acc, A2, lda2, K2, Wrow, K0 + K1,  arow, hi);

  const int n = tn * 16 + l15;
  const float bv = bias ? bias[n] : 0.0f;
#pragma unroll
  for (int r = 0; r < 8; ++r) {
    int m = tm * 16 + hi * 8 + r;
    if (m < M) C[(long long)m * ldc + (long long)n * cstride] = acc[r] + bv;
  }
}

// ---------------------------------------------------------------- fused LSTM step
// One wave owns one 16-column slice jt and ALL FOUR gate tiles: the A fragment
// is loaded once per K-chunk and feeds 4 WMMAs (4x less activation traffic).
// Gate g element r in this lane corresponds to (m = 8*hi + r, col = jt*16+l15)
// for every gate, so the LSTM pointwise update runs entirely in-registers:
// no LDS exchange, no barrier. h16_old is read by every block -> caller
// double-buffers h16_old vs h16_new.
__global__ __launch_bounds__(128) void lstm_step_fused(
    const _Float16* __restrict__ X0, int ldx0, int K0,
    const _Float16* __restrict__ X1, int ldx1, int K1,
    const _Float16* __restrict__ Hold, int ldh, int Kh,
    const _Float16* __restrict__ Wih, int ldw_ih,   // [4D x (K0+K1)]
    const _Float16* __restrict__ Whh, int ldw_hh,   // [4D x Kh]
    const float* __restrict__ bias,                 // [4D]
    float* __restrict__ c,                          // [16 x D] in/out
    _Float16* __restrict__ h16_new,                 // [16 x D] out
    int D)
{
  const int wave = threadIdx.x >> 5;
  const int lane = threadIdx.x & 31;
  const int jt   = blockIdx.x * 4 + wave;      // 16-column tile within D
  const int hi   = lane >> 4;
  const int l15  = lane & 15;
  const int col  = jt * 16 + l15;

  v8f g0 = {0.f,0.f,0.f,0.f,0.f,0.f,0.f,0.f};
  v8f g1 = g0, g2 = g0, g3 = g0;

  {
    const _Float16* w0 = Wih + (long long)(0 * D + col) * ldw_ih;
    const _Float16* w1 = Wih + (long long)(1 * D + col) * ldw_ih;
    const _Float16* w2 = Wih + (long long)(2 * D + col) * ldw_ih;
    const _Float16* w3 = Wih + (long long)(3 * D + col) * ldw_ih;
    wmma_seg4(g0, g1, g2, g3, X0, ldx0, K0, w0, w1, w2, w3, 0,  l15, hi);
    wmma_seg4(g0, g1, g2, g3, X1, ldx1, K1, w0, w1, w2, w3, K0, l15, hi);
  }
  {
    const _Float16* w0 = Whh + (long long)(0 * D + col) * ldw_hh;
    const _Float16* w1 = Whh + (long long)(1 * D + col) * ldw_hh;
    const _Float16* w2 = Whh + (long long)(2 * D + col) * ldw_hh;
    const _Float16* w3 = Whh + (long long)(3 * D + col) * ldw_hh;
    wmma_seg4(g0, g1, g2, g3, Hold, ldh, Kh, w0, w1, w2, w3, 0, l15, hi);
  }

  const float bi = bias[0 * D + col];
  const float bf = bias[1 * D + col];
  const float bg = bias[2 * D + col];
  const float bo = bias[3 * D + col];
#pragma unroll
  for (int r = 0; r < 8; ++r) {
    int m = hi * 8 + r;
    long long idx = (long long)m * D + col;
    float gi = g0[r] + bi, gf = g1[r] + bf, gg = g2[r] + bg, go = g3[r] + bo;
    float cv = sigmf(gf) * c[idx] + sigmf(gi) * tanhf(gg);
    float hv = sigmf(go) * tanhf(cv);
    c[idx] = cv;
    h16_new[idx] = (_Float16)hv;
  }
}

// ---------------------------------------------------------------- converters
__global__ void cvt_pad_f16(const float* __restrict__ src, _Float16* __restrict__ dst,
                            int rows, int K, int Kp)
{
  long long i = (long long)blockIdx.x * blockDim.x + threadIdx.x;
  long long total = (long long)rows * Kp;
  if (i >= total) return;
  int r = (int)(i / Kp), k = (int)(i % Kp);
  float v = (k < K) ? src[(long long)r * K + k] : 0.0f;
  dst[i] = (_Float16)v;
}

// Build prenet input: frames 0..200 (frame 0 = zeros), row = fi*16+b, Kpad=96
__global__ void build_prenet_in(const float* __restrict__ dec_inputs, _Float16* __restrict__ x0)
{
  int i = blockIdx.x * blockDim.x + threadIdx.x;
  const int total = (TDEC + 1) * B_ * 96;
  if (i >= total) return;
  int k = i % 96;
  int r = i / 96;
  int b = r % B_;
  int fi = r / B_;
  float v = 0.0f;
  if (k < NMELS && fi > 0)
    v = dec_inputs[((long long)b * NMELS + k) * TDEC + (fi - 1)];
  x0[i] = (_Float16)v;
}

// relu + deterministic hash dropout(p=0.5, x2 scale) + f16 quantize
__global__ void relu_drop_f16(const float* __restrict__ in, _Float16* __restrict__ out,
                              int n, unsigned seed)
{
  int i = blockIdx.x * blockDim.x + threadIdx.x;
  if (i >= n) return;
  float v = fmaxf(in[i], 0.0f);
  v = (hash_u32((unsigned)i * 2654435761u ^ seed) & 1u) ? v * 2.0f : 0.0f;
  out[i] = (_Float16)v;
}

__global__ void zero_f32(float* p, int n) {
  int i = blockIdx.x * blockDim.x + threadIdx.x;
  if (i < n) p[i] = 0.0f;
}
__global__ void zero_f16(_Float16* p, int n) {
  int i = blockIdx.x * blockDim.x + threadIdx.x;
  if (i < n) p[i] = (_Float16)0.0f;
}

// ---------------------------------------------------------------- attention step
// One block per batch element, one thread per encoder position t (256).
// Also computes query = ah @ query_w^T in-block (removes a 1-block launch
// from the critical path): 2 threads per query element, LDS combine.
__global__ __launch_bounds__(256) void attention_step(
    const _Float16* __restrict__ ah16,    // 16 x 1024 (current step)
    const _Float16* __restrict__ qw16,    // 128 x 1024 (f16 query weights)
    const float* __restrict__ pmem,       // 16 x 256 x 128
    const float* __restrict__ memory,     // 16 x 256 x 512
    const int*   __restrict__ memlen,     // 16
    const float* __restrict__ loc_conv_w, // 32 x 2 x 31
    const float* __restrict__ loc_dense_w,// 128 x 32
    const float* __restrict__ v_w,        // 128
    float* __restrict__ aw, float* __restrict__ awc,        // 16 x 256
    float* __restrict__ actx, _Float16* __restrict__ actx16)// 16 x 512
{
  __shared__ float s_cw[NFILT_ * 2 * KCONV];
  __shared__ float s_dw[ADIM_ * NFILT_];
  __shared__ float s_v[ADIM_];
  __shared__ float s_q[ADIM_];
  __shared__ float s_aw[TENC], s_awc[TENC], red[TENC];

  const int b = blockIdx.x, t = threadIdx.x;
  for (int i = t; i < NFILT_ * 2 * KCONV; i += 256) s_cw[i] = loc_conv_w[i];
  for (int i = t; i < ADIM_ * NFILT_;     i += 256) s_dw[i] = loc_dense_w[i];
  if (t < ADIM_) s_v[t] = v_w[t];
  s_aw[t]  = aw[b * TENC + t];
  s_awc[t] = awc[b * TENC + t];

  // query[b][a]: thread pair (2a, 2a+1) splits the 1024-long dot
  {
    const int a = t >> 1, hf = t & 1;
    const _Float16* hp = ah16 + (long long)b * ARNN_ + hf * 512;
    const _Float16* qr = qw16 + (long long)a * ARNN_ + hf * 512;
    float acc = 0.0f;
    for (int j = 0; j < 512; j += 8) {
      v8h hv = *(const v8h*)(hp + j);
      v8h qv = *(const v8h*)(qr + j);
#pragma unroll
      for (int u = 0; u < 8; ++u) acc += (float)hv[u] * (float)qv[u];
    }
    red[t] = acc;
  }
  __syncthreads();
  if (t < ADIM_) s_q[t] = red[2 * t] + red[2 * t + 1];
  __syncthreads();

  // location conv (cross-correlation, pad 15)
  float conv[NFILT_];
#pragma unroll
  for (int f = 0; f < NFILT_; ++f) conv[f] = 0.0f;
  for (int j = 0; j < KCONV; ++j) {
    int tt = t + j - (KCONV - 1) / 2;
    float x0 = 0.0f, x1 = 0.0f;
    if (tt >= 0 && tt < TENC) { x0 = s_aw[tt]; x1 = s_awc[tt]; }
#pragma unroll
    for (int f = 0; f < NFILT_; ++f)
      conv[f] += s_cw[f * 62 + j] * x0 + s_cw[f * 62 + KCONV + j] * x1;
  }

  // energies
  float score = 0.0f;
  const float* pm = pmem + ((long long)b * TENC + t) * ADIM_;
  for (int a = 0; a < ADIM_; ++a) {
    float loc = 0.0f;
#pragma unroll
    for (int f = 0; f < NFILT_; ++f) loc += s_dw[a * NFILT_ + f] * conv[f];
    float s = s_q[a] + loc + pm[a];
    s = fminf(20.0f, fmaxf(-20.0f, s));
    score += s_v[a] * tanhf(s);
  }
  if (t < memlen[b]) score = -__builtin_inff();   // bug-compatible with reference

  // softmax over T_ENC
  red[t] = score; __syncthreads();
  for (int off = 128; off > 0; off >>= 1) {
    if (t < off) red[t] = fmaxf(red[t], red[t + off]);
    __syncthreads();
  }
  float mx = red[0]; __syncthreads();
  float e = __expf(score - mx);
  red[t] = e; __syncthreads();
  for (int off = 128; off > 0; off >>= 1) {
    if (t < off) red[t] += red[t + off];
    __syncthreads();
  }
  float w = e / red[0];
  s_aw[t] = w;
  aw[b * TENC + t] = w;
  awc[b * TENC + t] = s_awc[t] + w;
  __syncthreads();

  // context: actx[b, e] = sum_t aw[t] * memory[b, t, e]
  for (int e0 = t; e0 < ENC_; e0 += 256) {
    float acc = 0.0f;
    const float* mb = memory + (long long)b * TENC * ENC_ + e0;
    for (int tt = 0; tt < TENC; ++tt) acc += s_aw[tt] * mb[(long long)tt * ENC_];
    actx[b * ENC_ + e0] = acc;
    actx16[b * ENC_ + e0] = (_Float16)acc;
  }
}

// ---------------------------------------------------------------- host side
static inline int ceil_div_ll(long long a, int b) { return (int)((a + b - 1) / b); }

static inline void launch_gemm(const _Float16* A0, int lda0, int K0,
                               const _Float16* A1, int lda1, int K1,
                               const _Float16* W, int ldw,
                               float* C, long long ldc, long long cstride,
                               int M, int N, const float* bias, hipStream_t s)
{
  int waves = ((M + 15) / 16) * (N / 16);
  int blocks = (waves + 7) / 8;
  gemm16_multi<<<blocks, 256, 0, s>>>(A0, lda0, K0, A1, lda1, K1,
                                      (const _Float16*)nullptr, 0, 0,
                                      W, ldw, C, ldc, cstride, M, N, bias);
}

static inline void launch_cvt(const float* src, _Float16* dst, int rows, int K, int Kp,
                              hipStream_t s)
{
  long long total = (long long)rows * Kp;
  cvt_pad_f16<<<ceil_div_ll(total, 256), 256, 0, s>>>(src, dst, rows, K, Kp);
}

static inline char* bump(char*& p, size_t bytes) {
  char* r = p;
  p += (bytes + 255) & ~(size_t)255;
  return r;
}

extern "C" void kernel_launch(void* const* d_in, const int* in_sizes, int n_in,
                              void* d_out, int out_size, void* d_ws, size_t ws_size,
                              hipStream_t stream)
{
  (void)in_sizes; (void)n_in; (void)out_size; (void)ws_size;

  const float* memory      = (const float*)d_in[0];
  const float* dec_inputs  = (const float*)d_in[1];
  const int*   memlen      = (const int*)  d_in[2];
  const float* prenet_w1   = (const float*)d_in[3];
  const float* prenet_w2   = (const float*)d_in[4];
  const float* attn_w_ih   = (const float*)d_in[5];
  const float* attn_w_hh   = (const float*)d_in[6];
  const float* attn_b      = (const float*)d_in[7];
  const float* dec_w_ih    = (const float*)d_in[8];
  const float* dec_w_hh    = (const float*)d_in[9];
  const float* dec_b       = (const float*)d_in[10];
  const float* proj_w      = (const float*)d_in[11];
  const float* proj_b      = (const float*)d_in[12];
  const float* query_w     = (const float*)d_in[13];
  const float* memory_w    = (const float*)d_in[14];
  const float* v_w         = (const float*)d_in[15];
  const float* loc_conv_w  = (const float*)d_in[16];
  const float* loc_dense_w = (const float*)d_in[17];
  float* out = (float*)d_out;                          // 16 x 80 x 200

  char* p = (char*)d_ws;

  // f16 weights (K padded to multiple of 32 where needed)
  _Float16* w_pre1  = (_Float16*)bump(p, (size_t)256  * 96   * 2);
  _Float16* w_pre2  = (_Float16*)bump(p, (size_t)256  * 256  * 2);
  _Float16* w_a_ih  = (_Float16*)bump(p, (size_t)4096 * 768  * 2);
  _Float16* w_a_hh  = (_Float16*)bump(p, (size_t)4096 * 1024 * 2);
  _Float16* w_d_ih  = (_Float16*)bump(p, (size_t)4096 * 1536 * 2);
  _Float16* w_d_hh  = (_Float16*)bump(p, (size_t)4096 * 1024 * 2);
  _Float16* w_proj  = (_Float16*)bump(p, (size_t)80   * 1536 * 2);
  _Float16* w_query = (_Float16*)bump(p, (size_t)128  * 1024 * 2);
  _Float16* w_memw  = (_Float16*)bump(p, (size_t)128  * 512  * 2);
  _Float16* mem16   = (_Float16*)bump(p, (size_t)B_ * TENC * ENC_ * 2);

  // prenet pipeline
  const int NPRE_ROWS = (TDEC + 1) * B_;               // 3216
  _Float16* x0       = (_Float16*)bump(p, (size_t)NPRE_ROWS * 96 * 2);
  float*    h1f32    = (float*)   bump(p, (size_t)NPRE_ROWS * PRE_ * 4);
  _Float16* h1f16    = (_Float16*)bump(p, (size_t)NPRE_ROWS * PRE_ * 2);
  float*    h2f32    = (float*)   bump(p, (size_t)NPRE_ROWS * PRE_ * 4);
  _Float16* decins16 = (_Float16*)bump(p, (size_t)NPRE_ROWS * PRE_ * 2);

  float* pmem = (float*)bump(p, (size_t)B_ * TENC * ADIM_ * 4);

  // recurrent state (c buffers f32; h in f16 double-buffered)
  const int NSTATE_F = B_*ARNN_ + B_*DEC_ + B_*ENC_ + B_*TENC*2;   // ac,dc,actx,aw,awc
  const int NSTATE_H = B_*ARNN_*2 + B_*DEC_*2 + B_*ENC_;           // ah16 x2, dh16 x2, actx16
  float*    stateF = (float*)   bump(p, (size_t)NSTATE_F * 4);
  _Float16* stateH = (_Float16*)bump(p, (size_t)NSTATE_H * 2);
  float* ac   = stateF;
  float* dc   = ac  + B_*ARNN_;
  float* actx = dc  + B_*DEC_;
  float* aw   = actx + B_*ENC_;
  float* awc  = aw   + B_*TENC;
  _Float16* ah16_a = stateH;
  _Float16* ah16_b = ah16_a + B_*ARNN_;
  _Float16* dh16_a = ah16_b + B_*ARNN_;
  _Float16* dh16_b = dh16_a + B_*DEC_;
  _Float16* actx16 = dh16_b + B_*DEC_;

  // ---- setup: weight conversion (fp32 -> f16, zero-padded K) ----
  launch_cvt(prenet_w1, w_pre1,  256,  80,   96,   stream);
  launch_cvt(prenet_w2, w_pre2,  256,  256,  256,  stream);
  launch_cvt(attn_w_ih, w_a_ih,  4096, 768,  768,  stream);
  launch_cvt(attn_w_hh, w_a_hh,  4096, 1024, 1024, stream);
  launch_cvt(dec_w_ih,  w_d_ih,  4096, 1536, 1536, stream);
  launch_cvt(dec_w_hh,  w_d_hh,  4096, 1024, 1024, stream);
  launch_cvt(proj_w,    w_proj,  80,   1536, 1536, stream);
  launch_cvt(query_w,   w_query, 128,  1024, 1024, stream);
  launch_cvt(memory_w,  w_memw,  128,  512,  512,  stream);
  launch_cvt(memory,    mem16,   B_ * TENC, ENC_, ENC_, stream);

  // ---- prenet over all 201 frames x 16 batch ----
  {
    int total = NPRE_ROWS * 96;
    build_prenet_in<<<ceil_div_ll(total, 256), 256, 0, stream>>>(dec_inputs, x0);
  }
  launch_gemm(x0, 96, 96, nullptr, 0, 0, w_pre1, 96,
              h1f32, PRE_, 1, NPRE_ROWS, PRE_, nullptr, stream);
  {
    int n = NPRE_ROWS * PRE_;
    relu_drop_f16<<<ceil_div_ll(n, 256), 256, 0, stream>>>(h1f32, h1f16, n, 0x9E3779B9u);
  }
  launch_gemm(h1f16, PRE_, PRE_, nullptr, 0, 0, w_pre2, PRE_,
              h2f32, PRE_, 1, NPRE_ROWS, PRE_, nullptr, stream);
  {
    int n = NPRE_ROWS * PRE_;
    relu_drop_f16<<<ceil_div_ll(n, 256), 256, 0, stream>>>(h2f32, decins16, n, 0x85EBCA77u);
  }

  // ---- pmem = memory @ memory_w^T ----
  launch_gemm(mem16, ENC_, ENC_, nullptr, 0, 0, w_memw, ENC_,
              pmem, ADIM_, 1, B_ * TENC, ADIM_, nullptr, stream);

  // ---- zero recurrent state ----
  zero_f32<<<ceil_div_ll(NSTATE_F, 256), 256, 0, stream>>>(stateF, NSTATE_F);
  zero_f16<<<ceil_div_ll(NSTATE_H, 256), 256, 0, stream>>>(stateH, NSTATE_H);

  // ---- autoregressive decode: 200 sequential steps, 4 launches each ----
  _Float16* ah16_old = ah16_a; _Float16* ah16_new = ah16_b;
  _Float16* dh16_old = dh16_a; _Float16* dh16_new = dh16_b;

  for (int t = 0; t < TDEC; ++t) {
    const _Float16* din16 = decins16 + (size_t)t * B_ * PRE_;

    // 1) attention LSTM: gates([dec_in|actx]) + recurrent + pointwise, fused
    lstm_step_fused<<<ARNN_ / 64, 128, 0, stream>>>(
        din16, PRE_, PRE_, actx16, ENC_, ENC_, ah16_old, ARNN_, ARNN_,
        w_a_ih, 768, w_a_hh, 1024, attn_b, ac, ah16_new, ARNN_);

    // 2) fused query + location-attention + softmax + context
    attention_step<<<B_, TENC, 0, stream>>>(ah16_new, w_query, pmem, memory, memlen,
                                            loc_conv_w, loc_dense_w, v_w,
                                            aw, awc, actx, actx16);

    // 3) decoder LSTM: gates([ah|actx]) + recurrent + pointwise, fused
    lstm_step_fused<<<DEC_ / 64, 128, 0, stream>>>(
        ah16_new, ARNN_, ARNN_, actx16, ENC_, ENC_, dh16_old, DEC_, DEC_,
        w_d_ih, 1536, w_d_hh, 1024, dec_b, dc, dh16_new, DEC_);

    // 4) mel projection: [dh|actx] @ proj_w^T + b -> out[b, mel, t]
    launch_gemm(dh16_new, DEC_, DEC_, actx16, ENC_, ENC_, w_proj, 1536,
                out + t, (long long)NMELS * TDEC, TDEC, B_, NMELS, proj_b, stream);

    // swap recurrent h double-buffers
    _Float16* tmp;
    tmp = ah16_old; ah16_old = ah16_new; ah16_new = tmp;
    tmp = dh16_old; dh16_old = dh16_new; dh16_new = tmp;
  }
}
